// InferenceRNN_34299608826603
// MI455X (gfx1250) — matrix operand order repaired and checked by reading, not compiled, not run on hardware
//
#include <hip/hip_runtime.h>
#include <hip/hip_bf16.h>

// Dims from reference
#define Z_   256
#define IN_  512
#define HH_  512
#define H_   1024
#define T_   512
#define B_   64
#define G4H  (4 * H_)   // 4096

typedef __attribute__((ext_vector_type(16))) __bf16 v16bf;
typedef __attribute__((ext_vector_type(8)))  float  v8f;

union Frag16 {
    v16bf v;
    uint4 q[2];
};

__device__ __forceinline__ unsigned short f2bf(float x) {
    unsigned u = __float_as_uint(x);
    return (unsigned short)((u + 0x7FFFu + ((u >> 16) & 1u)) >> 16);
}

__device__ __forceinline__ v8f wmma_bf16(const Frag16& a, const Frag16& b, v8f acc) {
    return __builtin_amdgcn_wmma_f32_16x16x32_bf16(
        /*neg_a=*/false, a.v, /*neg_b=*/false, b.v,
        /*c_mod=*/(short)0, acc, /*reuse_a=*/false, /*reuse_b=*/false);
}

// A fragment: 16x32 bf16, row-major source (K-contiguous rows).
// Lane layout (ISA 7.12.2): lanes 0-15 hold K=[0..7],[16..23]; lanes 16-31 hold K=[8..15],[24..31].
__device__ __forceinline__ void load_a_frag(Frag16& f, const unsigned short* rowBase,
                                            int k0, int grp) {
    const uint4* p = (const uint4*)(rowBase + k0 + grp * 8);
    f.q[0] = p[0];   // 8 elems
    f.q[1] = p[2];   // +16 elems
}

// B fragment: 32x16 bf16; lane n holds column n along K (source row n is K-contiguous).
// Lanes 0-15: K=[0..7],[16..23], lanes 16-31: K=[8..15],[24..31].
__device__ __forceinline__ void load_b_frag(Frag16& f, const unsigned short* colRowBase,
                                            int k0, int grp) {
    const uint4* p = (const uint4*)(colRowBase + k0 + grp * 8);
    f.q[0] = p[0];
    f.q[1] = p[2];
}

// ---------------------------------------------------------------- converts
__global__ void cvt_bf16_kernel(const float* __restrict__ src,
                                unsigned short* __restrict__ dst, int n) {
    for (int i = blockIdx.x * blockDim.x + threadIdx.x; i < n;
         i += gridDim.x * blockDim.x)
        dst[i] = f2bf(src[i]);
}

// ---------------------------------------------------------------- hypernet
// z[mat][b][zi] = h_hat[b,:] . w[mat][zi,:] + bias[mat][zi]   (mat = hh,hx,hb)
__global__ void hyper_z_kernel(const float* __restrict__ hhat,
                               const float* __restrict__ w0, const float* __restrict__ b0,
                               const float* __restrict__ w1, const float* __restrict__ b1,
                               const float* __restrict__ w2, const float* __restrict__ b2,
                               float* __restrict__ z) {
    int idx = blockIdx.x * blockDim.x + threadIdx.x;
    if (idx >= 3 * B_ * Z_) return;
    int mat = idx / (B_ * Z_);
    int r   = idx % (B_ * Z_);
    int b   = r >> 8;            // /Z_
    int zi  = r & (Z_ - 1);
    const float* w  = (mat == 0) ? w0 : (mat == 1) ? w1 : w2;
    const float* bb = (mat == 0) ? b0 : (mat == 1) ? b1 : b2;
    const float* hr = hhat + b * HH_;
    const float* wr = w + zi * HH_;
    float s = 0.f;
    for (int j = 0; j < HH_; ++j) s += hr[j] * wr[j];
    z[mat * B_ * Z_ + b * Z_ + zi] = s + bb[zi];
}

// dz[0]=d_z_h = z_h @ W_hz.T ; dz[1]=d_z_x = z_x @ W_xz.T ; dz[2]=b_z_b = z_b @ b_w.T + b_b
__global__ void hyper_d_kernel(const float* __restrict__ z,
                               const float* __restrict__ Whz,
                               const float* __restrict__ Wxz,
                               const float* __restrict__ bw,
                               const float* __restrict__ bb,
                               float* __restrict__ dz) {
    int idx = blockIdx.x * blockDim.x + threadIdx.x;
    if (idx >= 3 * B_ * G4H) return;
    int mat = idx / (B_ * G4H);
    int r   = idx % (B_ * G4H);
    int b   = r >> 12;           // /4096
    int k   = r & (G4H - 1);
    const float* w  = (mat == 0) ? Whz : (mat == 1) ? Wxz : bw;
    const float* zr = z + mat * B_ * Z_ + b * Z_;
    const float* wr = w + k * Z_;
    float s = 0.f;
    for (int j = 0; j < Z_; ++j) s += zr[j] * wr[j];
    if (mat == 2) s += bb[k];
    dz[mat * B_ * G4H + b * G4H + k] = s;
}

// Double-buffered 1Mx4N WMMA inner product over K (K % 64 == 0).
// acc[4] accumulates; aBase = A row base (K-contiguous); bBase[4] = B column-row bases.
template <int K>
__device__ __forceinline__ void gemm_1x4_db(v8f acc[4],
                                            const unsigned short* aBase,
                                            const unsigned short* const bBase[4],
                                            int grp) {
    Frag16 aA, aB;
    Frag16 bA[4], bB[4];
    load_a_frag(aA, aBase, 0, grp);
#pragma unroll
    for (int j = 0; j < 4; ++j) load_b_frag(bA[j], bBase[j], 0, grp);

    for (int k0 = 0; k0 < K; k0 += 64) {
        // stage B loads (k0+32) in flight while stage A computes
        load_a_frag(aB, aBase, k0 + 32, grp);
#pragma unroll
        for (int j = 0; j < 4; ++j) load_b_frag(bB[j], bBase[j], k0 + 32, grp);
        __builtin_prefetch(aBase + k0 + 96 < aBase + K ? aBase + k0 + 96 : aBase, 0, 3);
#pragma unroll
        for (int j = 0; j < 4; ++j) acc[j] = wmma_bf16(aA, bA[j], acc[j]);

        // stage A loads for k0+64 (clamped; redundant reload on last iter)
        int kn = (k0 + 64 < K) ? (k0 + 64) : (K - 32);
        load_a_frag(aA, aBase, kn, grp);
#pragma unroll
        for (int j = 0; j < 4; ++j) load_b_frag(bA[j], bBase[j], kn, grp);
#pragma unroll
        for (int j = 0; j < 4; ++j) acc[j] = wmma_bf16(aB, bB[j], acc[j]);
    }
}

// ---------------------------------------------------------------- big GEMM
// pre[row, n] = d_z_x[b,n] * (x[row,:] . Wx[n,:] + Wx_b[n]) + b_z_b[b,n]
// rows = T*B flattened (row = t*B + b), K = IN = 512.
// Each wave: 1 M tile x 4 N tiles (A reuse, 4 independent WMMA chains, double-buffered K).
__global__ void xw_wmma_kernel(const unsigned short* __restrict__ xbf,
                               const unsigned short* __restrict__ wxbf,
                               const float* __restrict__ wx_bias,
                               const float* __restrict__ dzx,
                               const float* __restrict__ bzb,
                               float* __restrict__ pre) {
    const int lane = threadIdx.x & 31;
    const int wid  = (blockIdx.x * blockDim.x + threadIdx.x) >> 5;
    const int mt   = wid >> 6;           // 2048 M tiles
    const int ng   = wid & 63;           // 64 N-groups of 4 tiles
    const int row0 = mt * 16, col0 = ng * 64;
    const int grp = lane >> 4, l15 = lane & 15;
    const unsigned short* aBase = xbf + (size_t)(row0 + l15) * IN_;
    const unsigned short* bBase[4];
#pragma unroll
    for (int j = 0; j < 4; ++j)
        bBase[j] = wxbf + (size_t)(col0 + j * 16 + l15) * IN_;

    v8f acc[4];
#pragma unroll
    for (int j = 0; j < 4; ++j) acc[j] = (v8f){0.f,0.f,0.f,0.f,0.f,0.f,0.f,0.f};

    gemm_1x4_db<IN_>(acc, aBase, bBase, grp);

#pragma unroll
    for (int j = 0; j < 4; ++j) {
#pragma unroll
        for (int r = 0; r < 8; ++r) {
            int row = row0 + r + grp * 8;
            int col = col0 + j * 16 + l15;
            int b   = row & (B_ - 1);
            float v = dzx[b * G4H + col] * (acc[j][r] + wx_bias[col]) + bzb[b * G4H + col];
            pre[(size_t)row * G4H + col] = v;
        }
    }
}

// ---------------------------------------------------------------- step GEMM
// ifgo[b, n] = d_z_h[b,n] * (h[b,:] . Wh[n,:] + Wh_b[n]) + pre_t[b,n]
// 4 M tiles x 256 N tiles; each wave: 1 M tile x 4 N tiles -> 256 waves (32 blocks).
__global__ void step_gemm_kernel(const unsigned short* __restrict__ hbf,
                                 const unsigned short* __restrict__ whbf,
                                 const float* __restrict__ wh_bias,
                                 const float* __restrict__ dzh,
                                 const float* __restrict__ pre_t,
                                 float* __restrict__ ifgo) {
    const int lane = threadIdx.x & 31;
    const int wid  = (blockIdx.x * blockDim.x + threadIdx.x) >> 5;
    const int mt   = wid >> 6;           // 4 M tiles
    const int ng   = wid & 63;           // 64 N-groups of 4 tiles
    const int row0 = mt * 16, col0 = ng * 64;
    const int grp = lane >> 4, l15 = lane & 15;
    const unsigned short* aBase = hbf + (size_t)(row0 + l15) * H_;
    const unsigned short* bBase[4];
#pragma unroll
    for (int j = 0; j < 4; ++j)
        bBase[j] = whbf + (size_t)(col0 + j * 16 + l15) * H_;

    v8f acc[4];
#pragma unroll
    for (int j = 0; j < 4; ++j) acc[j] = (v8f){0.f,0.f,0.f,0.f,0.f,0.f,0.f,0.f};

    gemm_1x4_db<H_>(acc, aBase, bBase, grp);

#pragma unroll
    for (int j = 0; j < 4; ++j) {
#pragma unroll
        for (int r = 0; r < 8; ++r) {
            int b   = row0 + r + grp * 8;    // batch row (0..63)
            int col = col0 + j * 16 + l15;
            float v = dzh[b * G4H + col] * (acc[j][r] + wh_bias[col]) + pre_t[b * G4H + col];
            ifgo[b * G4H + col] = v;
        }
    }
}

// ---------------------------------------------------------------- gates + LSTM + 2x LayerNorm
// One block per batch row; 256 threads x 4 elements.
__global__ void step_gate_kernel(const float* __restrict__ ifgo,
                                 float* __restrict__ c_state,
                                 unsigned short* __restrict__ hbf,
                                 float* __restrict__ out, int t,
                                 const float* __restrict__ lng, const float* __restrict__ lnb,
                                 const float* __restrict__ lcg, const float* __restrict__ lcb) {
    const int b   = blockIdx.x;
    const int tid = threadIdx.x;
    __shared__ float sh[4][256];
    float nc[4], nh[4];
    float s1 = 0.f, s2 = 0.f, s3 = 0.f, s4 = 0.f;
#pragma unroll
    for (int i = 0; i < 4; ++i) {
        int h    = tid + i * 256;
        float iv = ifgo[b * G4H + h];
        float fv = ifgo[b * G4H + H_ + h];
        float gv = ifgo[b * G4H + 2 * H_ + h];
        float ov = ifgo[b * G4H + 3 * H_ + h];
        float ig = 1.f / (1.f + __expf(-iv));
        float fg = 1.f / (1.f + __expf(-fv));
        float gg = tanhf(gv);
        float og = 1.f / (1.f + __expf(-ov));
        float c  = fg * c_state[b * H_ + h] + ig * gg;
        float hv = og * tanhf(c);        // pre-norm c, per reference
        nc[i] = c; nh[i] = hv;
        s1 += hv; s2 += hv * hv; s3 += c; s4 += c * c;
    }
    sh[0][tid] = s1; sh[1][tid] = s2; sh[2][tid] = s3; sh[3][tid] = s4;
    __syncthreads();
    for (int st = 128; st > 0; st >>= 1) {
        if (tid < st) {
            sh[0][tid] += sh[0][tid + st];
            sh[1][tid] += sh[1][tid + st];
            sh[2][tid] += sh[2][tid + st];
            sh[3][tid] += sh[3][tid + st];
        }
        __syncthreads();
    }
    const float inv = 1.0f / (float)H_;
    float mh = sh[0][0] * inv, vh = sh[1][0] * inv - mh * mh;
    float mc = sh[2][0] * inv, vc = sh[3][0] * inv - mc * mc;
    float rh = rsqrtf(vh + 1e-5f), rc = rsqrtf(vc + 1e-5f);
#pragma unroll
    for (int i = 0; i < 4; ++i) {
        int h = tid + i * 256;
        float nhv = (nh[i] - mh) * rh * lng[h] + lnb[h];
        float ncv = (nc[i] - mc) * rc * lcg[h] + lcb[h];
        c_state[b * H_ + h]  = ncv;
        hbf[b * H_ + h]      = f2bf(nhv);
        out[((size_t)b * T_ + t) * H_ + h] = nhv;   // out[B, T, H]
    }
}

// ---------------------------------------------------------------- launcher
extern "C" void kernel_launch(void* const* d_in, const int* in_sizes, int n_in,
                              void* d_out, int out_size, void* d_ws, size_t ws_size,
                              hipStream_t stream) {
    (void)in_sizes; (void)n_in; (void)out_size; (void)ws_size;

    const float* h0     = (const float*)d_in[0];
    const float* c0     = (const float*)d_in[1];
    const float* h_hat  = (const float*)d_in[2];
    const float* x_in   = (const float*)d_in[3];
    const float* w_hh_w = (const float*)d_in[4];
    const float* w_hh_b = (const float*)d_in[5];
    const float* w_hx_w = (const float*)d_in[6];
    const float* w_hx_b = (const float*)d_in[7];
    const float* w_hb_w = (const float*)d_in[8];
    const float* w_hb_b = (const float*)d_in[9];
    const float* W_hz_w = (const float*)d_in[10];
    const float* W_xz_w = (const float*)d_in[11];
    const float* b_w    = (const float*)d_in[12];
    const float* b_b    = (const float*)d_in[13];
    const float* Wh_w   = (const float*)d_in[14];
    const float* Wh_b   = (const float*)d_in[15];
    const float* Wx_w   = (const float*)d_in[16];
    const float* Wx_b   = (const float*)d_in[17];
    const float* ln_h_g = (const float*)d_in[18];
    const float* ln_h_b = (const float*)d_in[19];
    const float* ln_c_g = (const float*)d_in[20];
    const float* ln_c_b = (const float*)d_in[21];
    float* out = (float*)d_out;

    // workspace layout
    constexpr size_t SZ_PRE   = (size_t)T_ * B_ * G4H * sizeof(float);      // 512 MB
    constexpr size_t OFF_WHBF = SZ_PRE;
    constexpr size_t OFF_WXBF = OFF_WHBF + (size_t)G4H * H_ * 2;
    constexpr size_t OFF_XBF  = OFF_WXBF + (size_t)G4H * IN_ * 2;
    constexpr size_t OFF_Z    = OFF_XBF  + (size_t)T_ * B_ * IN_ * 2;
    constexpr size_t OFF_DZ   = OFF_Z    + (size_t)3 * B_ * Z_ * 4;
    constexpr size_t OFF_IFGO = OFF_DZ   + (size_t)3 * B_ * G4H * 4;
    constexpr size_t OFF_HBF  = OFF_IFGO + (size_t)B_ * G4H * 4;
    constexpr size_t OFF_C    = OFF_HBF  + (size_t)B_ * H_ * 2;

    char* ws = (char*)d_ws;
    float*          pre   = (float*)(ws);
    unsigned short* whbf  = (unsigned short*)(ws + OFF_WHBF);
    unsigned short* wxbf  = (unsigned short*)(ws + OFF_WXBF);
    unsigned short* xbf   = (unsigned short*)(ws + OFF_XBF);
    float*          zbuf  = (float*)(ws + OFF_Z);
    float*          dz    = (float*)(ws + OFF_DZ);
    float*          ifgo  = (float*)(ws + OFF_IFGO);
    unsigned short* hbf   = (unsigned short*)(ws + OFF_HBF);
    float*          c_st  = (float*)(ws + OFF_C);

    const float* dzh = dz;
    const float* dzx = dz + (size_t)B_ * G4H;
    const float* bzb = dz + (size_t)2 * B_ * G4H;

    // 1) bf16 conversions (Wh, Wx, x, h0)
    cvt_bf16_kernel<<<4096, 256, 0, stream>>>(Wh_w, whbf, G4H * H_);
    cvt_bf16_kernel<<<4096, 256, 0, stream>>>(Wx_w, wxbf, G4H * IN_);
    cvt_bf16_kernel<<<4096, 256, 0, stream>>>(x_in, xbf, T_ * B_ * IN_);
    cvt_bf16_kernel<<<256,  256, 0, stream>>>(h0, hbf, B_ * H_);
    hipMemcpyAsync(c_st, c0, (size_t)B_ * H_ * sizeof(float),
                   hipMemcpyDeviceToDevice, stream);

    // 2) hypernetwork (tiny)
    hyper_z_kernel<<<(3 * B_ * Z_ + 255) / 256, 256, 0, stream>>>(
        h_hat, w_hh_w, w_hh_b, w_hx_w, w_hx_b, w_hb_w, w_hb_b, zbuf);
    hyper_d_kernel<<<(3 * B_ * G4H + 255) / 256, 256, 0, stream>>>(
        zbuf, W_hz_w, W_xz_w, b_w, b_b, dz);

    // 3) input projection + hyper-scaling, all timesteps (WMMA, parallel over T)
    //    waves: 2048 M-tiles * 64 N-groups = 131072 -> 16384 blocks of 8 waves
    xw_wmma_kernel<<<16384, 256, 0, stream>>>(xbf, wxbf, Wx_b, dzx, bzb, pre);

    // 4) sequential recurrence: per step one WMMA GEMM + fused gate/LN kernel
    for (int t = 0; t < T_; ++t) {
        step_gemm_kernel<<<32, 256, 0, stream>>>(
            hbf, whbf, Wh_b, dzh, pre + (size_t)t * B_ * G4H, ifgo);
        step_gate_kernel<<<B_, 256, 0, stream>>>(
            ifgo, c_st, hbf, out, t, ln_h_g, ln_h_b, ln_c_g, ln_c_b);
    }
}
